// TurboQuantMSE_2860448219958
// MI455X (gfx1250) — compile-verified, matrix operand order checked by reading, not run on hardware
//
#include <hip/hip_runtime.h>

// ---------------------------------------------------------------------------
// TurboQuant MSE 4-bit: x_hat = Q(x @ R^T) @ R
//   - two 2048x4096x4096 GEMMs -> f16 WMMA (v_wmma_f32_16x16x32_f16)
//   - double-buffered GLOBAL_LOAD_ASYNC_TO_LDS_B128 staging (ASYNCcnt) when
//     the toolchain exposes the gfx1250 async builtins; sync fallback else
//   - quantization fused into GEMM1 epilogue (16-way nearest-centroid scan)
// ---------------------------------------------------------------------------

typedef __attribute__((ext_vector_type(8)))  _Float16 v8h;
typedef __attribute__((ext_vector_type(16))) _Float16 v16h;
typedef __attribute__((ext_vector_type(8)))  float    v8f;

#define DIM 4096
#define BM  128       // block tile M
#define BN  128       // block tile N
#define BK  32        // K step = native f16 WMMA K
#define LDT 40        // padded LDS row stride in halves (80B, 16B-aligned)

#if defined(__has_builtin)
#if __has_builtin(__builtin_amdgcn_global_load_async_to_lds_b128)
#define HAVE_ASYNC_LDS 1
#endif
#endif
#ifndef HAVE_ASYNC_LDS
#define HAVE_ASYNC_LDS 0
#endif

#if HAVE_ASYNC_LDS
// Builtin signature (from clang diagnostic): param 0 is
// 'int __vector(4) __device__ *'  ==  addrspace(1) pointer to 16B int vector.
typedef int v4i __attribute__((vector_size(16)));
typedef __attribute__((address_space(1))) v4i g_v4i;   // global
typedef __attribute__((address_space(3))) v4i l_v4i;   // LDS
__device__ __forceinline__ g_v4i* to_glb(const void* p) {
  return (g_v4i*)const_cast<void*>(p);
}
__device__ __forceinline__ l_v4i* to_lds(void* p) { return (l_v4i*)p; }
#define WAIT_ASYNC(n) asm volatile("s_wait_asynccnt %0" ::"n"(n) : "memory")
#endif

__device__ __forceinline__ v16h combine16(v8h lo, v8h hi) {
  v16h r;
#pragma unroll
  for (int i = 0; i < 8; ++i) { r[i] = lo[i]; r[i + 8] = hi[i]; }
  return r;
}

// ---------------- f32 -> f16 elementwise convert (x) ------------------------
__global__ __launch_bounds__(256) void convert_x_kernel(
    const float* __restrict__ x, _Float16* __restrict__ xh, int n) {
  int i = (blockIdx.x * 256 + threadIdx.x) * 8;
  if (i >= n) return;
  v8h o;
#pragma unroll
  for (int k = 0; k < 8; ++k) o[k] = (_Float16)x[i + k];
  *(v8h*)(xh + i) = o;
}

// ------------- R -> f16 copy + f16 transpose (LDS-tiled) --------------------
__global__ __launch_bounds__(256) void convert_rot_kernel(
    const float* __restrict__ R, _Float16* __restrict__ Rh,
    _Float16* __restrict__ RhT) {
  __shared__ _Float16 tile[32][33];
  const int bx = blockIdx.x * 32, by = blockIdx.y * 32;
  const int tx = threadIdx.x, ty = threadIdx.y;   // block = (32, 8)
#pragma unroll
  for (int r = 0; r < 4; ++r) {
    int row = by + ty + r * 8;
    _Float16 v = (_Float16)R[(size_t)row * DIM + bx + tx];
    Rh[(size_t)row * DIM + bx + tx] = v;          // straight f16 copy
    tile[ty + r * 8][tx] = v;
  }
  __syncthreads();
#pragma unroll
  for (int r = 0; r < 4; ++r) {
    int row = bx + ty + r * 8;                    // transposed coordinates
    RhT[(size_t)row * DIM + by + tx] = tile[tx][ty + r * 8];
  }
}

// ---------------- WMMA GEMM: Out = A (MxK) * Bt^T (KxN) ---------------------
// A row-major f16, Bt row-major f16 holding B^T (N x K).
// QUANT=true: epilogue quantizes to nearest codebook level, writes f16.
// QUANT=false: writes raw f32.
template <bool QUANT>
__global__ __launch_bounds__(256) void gemm_kernel(
    const _Float16* __restrict__ A, const _Float16* __restrict__ Bt,
    void* __restrict__ Out, const float* __restrict__ codebook, int K) {
  // Two stages, each: A tile (BM x BK, padded) then B tile (BN x BK, padded).
  __shared__ _Float16 lds[2][(BM + BN) * LDT];

  const int tid   = threadIdx.x;
  const int lane  = tid & 31;
  const int wave  = tid >> 5;        // 8 waves
  const int waveM = wave & 3;        // 4 waves along M, 32 rows each
  const int waveN = wave >> 2;       // 2 waves along N, 64 cols each
  const int blockM = blockIdx.y * BM;
  const int blockN = blockIdx.x * BN;

  float cb[16];
  if constexpr (QUANT) {
#pragma unroll
    for (int i = 0; i < 16; ++i) cb[i] = codebook[i];
  }

  v8f zero = {};
  v8f acc[2][4];
#pragma unroll
  for (int i = 0; i < 2; ++i)
#pragma unroll
    for (int j = 0; j < 4; ++j) acc[i][j] = zero;

  const int hl = lane & 15;
  const int kb = (lane < 16) ? 0 : 8;   // K-base of this half-wave's A/B frag

  // Global->LDS tiling: 128 rows x 32 halves = 512 chunks of 8 halves (16B);
  // 256 threads x 2 chunks each, for A and for B: 4 transfers/thread/tile.
  const int c0 = tid,       r0 = c0 >> 2, k0 = (c0 & 3) * 8;
  const int c1 = tid + 256, r1 = c1 >> 2, k1 = (c1 & 3) * 8;

  const _Float16* gA0 = &A[(size_t)(blockM + r0) * K + k0];
  const _Float16* gA1 = &A[(size_t)(blockM + r1) * K + k1];
  const _Float16* gB0 = &Bt[(size_t)(blockN + r0) * K + k0];
  const _Float16* gB1 = &Bt[(size_t)(blockN + r1) * K + k1];

  auto issue_tile = [&](int kt, int s) {
    _Float16* sA = &lds[s][0];
    _Float16* sB = &lds[s][BM * LDT];
#if HAVE_ASYNC_LDS
    __builtin_amdgcn_global_load_async_to_lds_b128(
        to_glb(gA0 + kt), to_lds(&sA[r0 * LDT + k0]), 0, 0);
    __builtin_amdgcn_global_load_async_to_lds_b128(
        to_glb(gA1 + kt), to_lds(&sA[r1 * LDT + k1]), 0, 0);
    __builtin_amdgcn_global_load_async_to_lds_b128(
        to_glb(gB0 + kt), to_lds(&sB[r0 * LDT + k0]), 0, 0);
    __builtin_amdgcn_global_load_async_to_lds_b128(
        to_glb(gB1 + kt), to_lds(&sB[r1 * LDT + k1]), 0, 0);
#else
    *(v8h*)&sA[r0 * LDT + k0] = *(const v8h*)(gA0 + kt);
    *(v8h*)&sA[r1 * LDT + k1] = *(const v8h*)(gA1 + kt);
    *(v8h*)&sB[r0 * LDT + k0] = *(const v8h*)(gB0 + kt);
    *(v8h*)&sB[r1 * LDT + k1] = *(const v8h*)(gB1 + kt);
#endif
  };

  const int nt = K / BK;
  issue_tile(0, 0);                       // prologue: tile 0 -> stage 0

  for (int t = 0; t < nt; ++t) {
    const int s = t & 1;
    if (t + 1 < nt) {
      issue_tile((t + 1) * BK, s ^ 1);    // stream next tile into other stage
      __builtin_prefetch(gA0 + (t + 2) * BK, 0, 1);  // nudge tile t+2 to L2
      __builtin_prefetch(gB0 + (t + 2) * BK, 0, 1);
#if HAVE_ASYNC_LDS
      WAIT_ASYNC(4);                      // in-order: tile t's 4 loads landed
#endif
    } else {
#if HAVE_ASYNC_LDS
      WAIT_ASYNC(0);
#endif
    }
    __syncthreads();                      // all waves' tile-t data visible

    const _Float16* sA = &lds[s][0];
    const _Float16* sB = &lds[s][BM * LDT];
    v16h af[2], bf[4];
#pragma unroll
    for (int i = 0; i < 2; ++i) {
      int row = waveM * 32 + i * 16 + hl;
      af[i] = combine16(*(const v8h*)&sA[row * LDT + kb],
                        *(const v8h*)&sA[row * LDT + kb + 16]);
    }
#pragma unroll
    for (int j = 0; j < 4; ++j) {
      int col = waveN * 64 + j * 16 + hl;
      bf[j] = combine16(*(const v8h*)&sB[col * LDT + kb],
                        *(const v8h*)&sB[col * LDT + kb + 16]);
    }

#pragma unroll
    for (int i = 0; i < 2; ++i)
#pragma unroll
      for (int j = 0; j < 4; ++j)
        acc[i][j] = __builtin_amdgcn_wmma_f32_16x16x32_f16(
            /*neg_a=*/false, af[i], /*neg_b=*/false, bf[j],
            /*c_mod=*/(short)0, acc[i][j],
            /*reuse_a=*/false, /*reuse_b=*/false);

    __syncthreads();                      // tile-t buffers safe to overwrite
  }

  // Epilogue. C/D layout: lanes 0-15 -> M = g, lanes 16-31 -> M = g + 8,
  // N = lane & 15, g = accumulator VGPR index.
#pragma unroll
  for (int i = 0; i < 2; ++i) {
#pragma unroll
    for (int j = 0; j < 4; ++j) {
      const int col   = blockN + waveN * 64 + j * 16 + hl;
      const int rbase = blockM + waveM * 32 + i * 16 + ((lane >> 4) << 3);
#pragma unroll
      for (int g = 0; g < 8; ++g) {
        float v = acc[i][j][g];
        if constexpr (QUANT) {
          // y_norm = y / SCALE = y * sqrt(4096); nearest of 16 Lloyd-Max levels
          float yn = v * 64.0f;
          float best = cb[0];
          float bd = __builtin_fabsf(yn - cb[0]);
#pragma unroll
          for (int q = 1; q < 16; ++q) {
            float d = __builtin_fabsf(yn - cb[q]);
            if (d < bd) { bd = d; best = cb[q]; }
          }
          ((_Float16*)Out)[(size_t)(rbase + g) * DIM + col] =
              (_Float16)(best * 0.015625f);          // * SCALE
        } else {
          ((float*)Out)[(size_t)(rbase + g) * DIM + col] = v;
        }
      }
    }
  }
}

// ---------------------------------------------------------------------------
extern "C" void kernel_launch(void* const* d_in, const int* in_sizes, int n_in,
                              void* d_out, int out_size, void* d_ws,
                              size_t ws_size, hipStream_t stream) {
  const float* x  = (const float*)d_in[0];
  const float* R  = (const float*)d_in[1];
  const float* cb = (const float*)d_in[2];
  const int rows = in_sizes[0] / DIM;        // 2048

  // Workspace layout (f16): Rh[DIM*DIM] | RhT[DIM*DIM] | xh[rows*DIM] | yh[rows*DIM]
  char* ws = (char*)d_ws;
  _Float16* Rh  = (_Float16*)(ws);
  _Float16* RhT = (_Float16*)(ws + (size_t)DIM * DIM * 2);
  _Float16* xh  = (_Float16*)(ws + (size_t)DIM * DIM * 4);
  _Float16* yh  = (_Float16*)(ws + (size_t)DIM * DIM * 4 + (size_t)rows * DIM * 2);

  const int nx = rows * DIM;
  convert_x_kernel<<<(nx + 2047) / 2048, 256, 0, stream>>>(x, xh, nx);
  convert_rot_kernel<<<dim3(DIM / 32, DIM / 32), dim3(32, 8), 0, stream>>>(R, Rh, RhT);

  dim3 grid(DIM / BN, rows / BM);
  // GEMM1 (fused quantize): y_hat = Q(x @ R^T); Bt = R itself (Bt[n][k] = R[n][k])
  gemm_kernel<true><<<grid, 256, 0, stream>>>(xh, Rh, yh, cb, DIM);
  // GEMM2: x_hat = y_hat @ R; Bt = R^T (Bt[n][k] = R[k][n])
  gemm_kernel<false><<<grid, 256, 0, stream>>>(yh, RhT, d_out, cb, DIM);
}
